// AttentiveVisitPooling_77077483094361
// MI455X (gfx1250) — matrix operand order repaired and checked by reading, not compiled, not run on hardware
//
#include <hip/hip_runtime.h>
#include <hip/hip_bf16.h>
#include <math.h>

typedef __attribute__((ext_vector_type(16))) __bf16 v16bf;
typedef __attribute__((ext_vector_type(8)))  float  v8f;

#define N_NODES 20000
#define E_VIS   1024
#define DIM     256
#define LN_EPS  1e-5f

__device__ __forceinline__ __bf16 f2bf(float f) { return (__bf16)f; }

__device__ __forceinline__ v8f v8f_zero() {
    v8f r;
    #pragma unroll
    for (int i = 0; i < 8; ++i) r[i] = 0.0f;
    return r;
}

// ---------------------------------------------------------------------------
// K0: zero accumulators
// ---------------------------------------------------------------------------
__global__ void k_zero(float* __restrict__ p, int n) {
    int i = blockIdx.x * 256 + threadIdx.x;
    if (i < n) p[i] = 0.0f;
}

// ---------------------------------------------------------------------------
// K1: s[n] = tanh(X @ W^T + b + q) . v   via WMMA bf16
// One wave per 16-row tile of X; K loop over D=256 in steps of 32.
// ---------------------------------------------------------------------------
__global__ void __launch_bounds__(256)
k_scores(const float* __restrict__ X, const float* __restrict__ W,
         const float* __restrict__ b, const float* __restrict__ q,
         const float* __restrict__ v, float* __restrict__ s)
{
    const int wave = threadIdx.x >> 5;
    const int lane = threadIdx.x & 31;
    const int h = lane >> 4, lm = lane & 15;
    const int tile = blockIdx.x * 8 + wave;
    if (tile >= N_NODES / 16) return;          // whole-wave exit, no barriers here
    const int row0 = tile * 16;

    v8f C[16];
    #pragma unroll
    for (int jt = 0; jt < 16; ++jt) C[jt] = v8f_zero();

    const float* Xrow = X + (size_t)(row0 + lm) * DIM;

    #pragma unroll 1
    for (int ks = 0; ks < 8; ++ks) {
        const int d0 = ks * 32;
        // A fragment (16x32 bf16): lane half h -> K = d0 + 16h + e
        v16bf a;
        #pragma unroll
        for (int ei = 0; ei < 16; ++ei)
            a[ei] = f2bf(Xrow[d0 + 16 * h + ei]);
        #pragma unroll
        for (int jt = 0; jt < 16; ++jt) {
            // B fragment (32x16 bf16): B[k][j] = W[j][k]; K = d0 + 16h + e, col = lm
            const float* Wrow = W + (size_t)(jt * 16 + lm) * DIM + d0 + 16 * h;
            v16bf bb;
            #pragma unroll
            for (int ei = 0; ei < 16; ++ei) bb[ei] = f2bf(Wrow[ei]);
            C[jt] = __builtin_amdgcn_wmma_f32_16x16x32_bf16(
                false, a, false, bb, (short)0, C[jt], false, false);
        }
    }

    // epilogue: per-lane partial of sum_j tanh(z)*v[j], rows r+8h, col j=jt*16+lm
    float sacc[8];
    #pragma unroll
    for (int r = 0; r < 8; ++r) sacc[r] = 0.0f;
    #pragma unroll
    for (int jt = 0; jt < 16; ++jt) {
        const int j = jt * 16 + lm;
        const float bq = b[j] + q[j];
        const float vv = v[j];
        #pragma unroll
        for (int r = 0; r < 8; ++r) sacc[r] += tanhf(C[jt][r] + bq) * vv;
    }
    // butterfly sum across the 16 lanes of each half (masks < 16 stay in-half)
    #pragma unroll
    for (int off = 1; off < 16; off <<= 1) {
        #pragma unroll
        for (int r = 0; r < 8; ++r) sacc[r] += __shfl_xor(sacc[r], off, 32);
    }
    if (lm == 0) {
        #pragma unroll
        for (int r = 0; r < 8; ++r) s[row0 + 8 * h + r] = sacc[r];
    }
}

// ---------------------------------------------------------------------------
// K2: smax = max(s)   (single 1024-thread block)
// ---------------------------------------------------------------------------
__global__ void k_maxreduce(const float* __restrict__ s, float* __restrict__ smax)
{
    __shared__ float red[1024];
    float m = -INFINITY;
    for (int i = threadIdx.x; i < N_NODES; i += 1024) m = fmaxf(m, s[i]);
    red[threadIdx.x] = m;
    __syncthreads();
    for (int off = 512; off > 0; off >>= 1) {
        if (threadIdx.x < off)
            red[threadIdx.x] = fmaxf(red[threadIdx.x], red[threadIdx.x + off]);
        __syncthreads();
    }
    if (threadIdx.x == 0) smax[0] = red[0];
}

// ---------------------------------------------------------------------------
// K3: t[n] = exp(s[n] - smax)
// ---------------------------------------------------------------------------
__global__ void k_expt(const float* __restrict__ s, const float* __restrict__ smax,
                       float* __restrict__ t)
{
    int i = blockIdx.x * 256 + threadIdx.x;
    if (i < N_NODES) t[i] = expf(s[i] - smax[0]);
}

// ---------------------------------------------------------------------------
// K4: pooled[e][j] += sum_n H[n][e]*t[n]*X[n][j] ; denom[e] += sum_n H[n][e]*t[n]
// Split-K WMMA GEMM. Block = 8 waves; covers 128 e x 256 j; grid (8, 25).
// Chunk = 800 n's = 25 K-iters of 32 (exact, no guards).
// ---------------------------------------------------------------------------
__global__ void __launch_bounds__(256)
k_pool_gemm(const float* __restrict__ H, const float* __restrict__ X,
            const float* __restrict__ t,
            float* __restrict__ pooled, float* __restrict__ denom)
{
    __shared__ __bf16 lds_xt[DIM][34];    // [j][k]  X^T tile (bf16), padded
    __shared__ __bf16 lds_hat[128][34];   // [e][k]  (H*t)^T tile (bf16), padded

    const int tid  = threadIdx.x;
    const int wave = tid >> 5, lane = tid & 31;
    const int h = lane >> 4, lm = lane & 15;
    const int e0 = blockIdx.x * 128;
    const int chunk0 = blockIdx.y * 800;

    v8f C[16];
    #pragma unroll
    for (int jt = 0; jt < 16; ++jt) C[jt] = v8f_zero();
    float dacc = 0.0f;

    #pragma unroll 1
    for (int kt = 0; kt < 25; ++kt) {
        const int n0 = chunk0 + kt * 32;
        __syncthreads();
        // stage X^T: thread tid owns column j=tid; coalesced row reads
        #pragma unroll 4
        for (int i = 0; i < 32; ++i)
            lds_xt[tid][i] = f2bf(X[(size_t)(n0 + i) * DIM + tid]);
        // stage (H*t)^T: 32 rows x 128 cols, coalesced 512B runs
        #pragma unroll 4
        for (int i = 0; i < 16; ++i) {
            const int idx = tid + 256 * i;
            const int row = idx >> 7, col = idx & 127;
            lds_hat[col][row] =
                f2bf(H[(size_t)(n0 + row) * E_VIS + e0 + col] * t[n0 + row]);
        }
        __syncthreads();

        // denom partial from the same bf16 A values (consistent with matmul)
        if (tid < 128) {
            #pragma unroll 8
            for (int k = 0; k < 32; ++k) dacc += (float)lds_hat[tid][k];
        }
        // A fragment: rows e = e0 + wave*16 + lm; K = 16h + e
        v16bf a;
        #pragma unroll
        for (int ei = 0; ei < 16; ++ei)
            a[ei] = lds_hat[wave * 16 + lm][16 * h + ei];
        #pragma unroll
        for (int jt = 0; jt < 16; ++jt) {
            // B fragment: col j = jt*16+lm ; K = 16h + e (contiguous 16 bf16)
            v16bf bb;
            #pragma unroll
            for (int ei = 0; ei < 16; ++ei)
                bb[ei] = lds_xt[jt * 16 + lm][16 * h + ei];
            C[jt] = __builtin_amdgcn_wmma_f32_16x16x32_bf16(
                false, a, false, bb, (short)0, C[jt], false, false);
        }
    }

    // split-K accumulation
    #pragma unroll
    for (int jt = 0; jt < 16; ++jt) {
        #pragma unroll
        for (int r = 0; r < 8; ++r)
            atomicAdd(&pooled[(size_t)(e0 + wave * 16 + r + 8 * h) * DIM +
                              jt * 16 + lm], C[jt][r]);
    }
    if (tid < 128) atomicAdd(&denom[e0 + tid], dacc);
}

// ---------------------------------------------------------------------------
// K5: per-visit LayerNorm of pooled/denom -> Vn
// ---------------------------------------------------------------------------
__global__ void k_layernorm(const float* __restrict__ pooled,
                            const float* __restrict__ denom,
                            const float* __restrict__ gamma,
                            const float* __restrict__ beta,
                            float* __restrict__ Vn)
{
    __shared__ float red[DIM];
    const int e = blockIdx.x, j = threadIdx.x;
    const float p = pooled[(size_t)e * DIM + j] / fmaxf(denom[e], 1e-30f);
    red[j] = p;
    __syncthreads();
    for (int off = 128; off > 0; off >>= 1) {
        if (j < off) red[j] += red[j + off];
        __syncthreads();
    }
    const float mu = red[0] * (1.0f / DIM);
    __syncthreads();
    const float d = p - mu;
    red[j] = d * d;
    __syncthreads();
    for (int off = 128; off > 0; off >>= 1) {
        if (j < off) red[j] += red[j + off];
        __syncthreads();
    }
    const float var = red[0] * (1.0f / DIM);
    Vn[(size_t)e * DIM + j] = d * rsqrtf(var + LN_EPS) * gamma[j] + beta[j];
}

// ---------------------------------------------------------------------------
// K6: out[j] = max over e of Vn[e][j]
// ---------------------------------------------------------------------------
__global__ void k_colmax(const float* __restrict__ Vn, float* __restrict__ out)
{
    const int j = threadIdx.x;
    float m = -INFINITY;
    #pragma unroll 4
    for (int e = 0; e < E_VIS; ++e) m = fmaxf(m, Vn[(size_t)e * DIM + j]);
    out[j] = m;
}

// ---------------------------------------------------------------------------
extern "C" void kernel_launch(void* const* d_in, const int* in_sizes, int n_in,
                              void* d_out, int out_size, void* d_ws, size_t ws_size,
                              hipStream_t stream)
{
    const float* X  = (const float*)d_in[0];   // [N, D]
    const float* H  = (const float*)d_in[1];   // [N, E]
    const float* W  = (const float*)d_in[2];   // [D, D]
    const float* b  = (const float*)d_in[3];   // [D]
    const float* v  = (const float*)d_in[4];   // [D]
    const float* q  = (const float*)d_in[5];   // [D]
    const float* g  = (const float*)d_in[6];   // ln_gamma [D]
    const float* be = (const float*)d_in[7];   // ln_beta  [D]
    float* out = (float*)d_out;                // [D]

    float* ws     = (float*)d_ws;
    float* s_buf  = ws;                        // 20000
    float* t_buf  = s_buf + N_NODES;           // 20000
    float* smax   = t_buf + N_NODES;           // 32 (padded)
    float* denom  = smax + 32;                 // 1024
    float* pooled = denom + E_VIS;             // 1024*256
    float* Vn     = pooled + (size_t)E_VIS * DIM; // 1024*256
    (void)in_sizes; (void)n_in; (void)out_size; (void)ws_size;

    const int nz = E_VIS + E_VIS * DIM;        // denom + pooled are contiguous
    k_zero<<<(nz + 255) / 256, 256, 0, stream>>>(denom, nz);
    k_scores<<<(N_NODES / 16 + 7) / 8, 256, 0, stream>>>(X, W, b, q, v, s_buf);
    k_maxreduce<<<1, 1024, 0, stream>>>(s_buf, smax);
    k_expt<<<(N_NODES + 255) / 256, 256, 0, stream>>>(s_buf, smax, t_buf);
    k_pool_gemm<<<dim3(E_VIS / 128, 25), 256, 0, stream>>>(H, X, t_buf, pooled, denom);
    k_layernorm<<<E_VIS, DIM, 0, stream>>>(pooled, denom, g, be, Vn);
    k_colmax<<<1, DIM, 0, stream>>>(Vn, out);
}
